// Constellation_CNN_80917183857212
// MI455X (gfx1250) — compile-verified, matrix operand order checked by reading, not run on hardware
//
#include <hip/hip_runtime.h>
#include <hip/hip_bf16.h>

typedef __attribute__((ext_vector_type(16))) _Float16 v16h;
typedef __attribute__((ext_vector_type(8)))  float    v8f;

union Frag {
    v16h h;
    unsigned u[8];
};

// ---------------------------------------------------------------------------
// Histogram + threshold -> 3 indicator channels, NHWC f16 (C padded 3->4)
// d0[((b*32+y)*32+x)*4 + c], c = {high, med, low, 0}
// ---------------------------------------------------------------------------
__global__ __launch_bounds__(256) void hist_kernel(const float* __restrict__ x,
                                                   _Float16* __restrict__ d0) {
    __shared__ unsigned hist[1024];
    const int b = blockIdx.x, tid = threadIdx.x;
    for (int i = tid; i < 1024; i += 256) hist[i] = 0u;
    __syncthreads();
    const float* xi = x + (size_t)b * 2048;
    const float* xq = xi + 1024;
    for (int j = tid; j < 1024; j += 256) {
        float iv = xi[j], qv = xq[j];
        bool valid = (iv >= -2.f) && (iv <= 2.f) && (qv >= -2.f) && (qv <= 2.f);
        if (valid) {
            int ii = (int)floorf((iv + 2.f) * 8.f);
            int qq = (int)floorf((qv + 2.f) * 8.f);
            ii = min(max(ii, 0), 31);
            qq = min(max(qq, 0), 31);
            atomicAdd(&hist[ii * 32 + qq], 1u);
        }
    }
    __syncthreads();
    for (int i = tid; i < 1024; i += 256) {
        float H = (float)hist[i];
        _Float16 hi = (H > 15.f) ? (_Float16)1.f : (_Float16)0.f;
        _Float16 me = (H > 5.f && H <= 15.f) ? (_Float16)1.f : (_Float16)0.f;
        _Float16 lo = (H <= 5.f) ? (_Float16)1.f : (_Float16)0.f;
        _Float16* p = d0 + ((size_t)b * 1024 + i) * 4;
        p[0] = hi; p[1] = me; p[2] = lo; p[3] = (_Float16)0.f;
    }
}

// ---------------------------------------------------------------------------
// Weight repack: cw[COUT][CIN][KS][KS] f32 -> Wt[n][Kpad] f16, k = s*CINP + c
// ---------------------------------------------------------------------------
template <int COUT, int CIN, int CINP, int KS, int KPAD>
__global__ __launch_bounds__(256) void repack_conv(const float* __restrict__ w,
                                                   _Float16* __restrict__ Wt) {
    int i = blockIdx.x * 256 + threadIdx.x;
    if (i >= COUT * KPAD) return;
    int n = i / KPAD, k = i % KPAD;
    int s = k / CINP, c = k % CINP;
    float v = 0.f;
    if (s < KS * KS && c < CIN) {
        int ky = s / KS, kx = s % KS;
        v = w[(((size_t)n * CIN + c) * KS + ky) * KS + kx];
    }
    Wt[i] = (_Float16)v;
}

__global__ __launch_bounds__(256) void cast_f32_f16(const float* __restrict__ w,
                                                    _Float16* __restrict__ o, int n) {
    int i = blockIdx.x * 256 + threadIdx.x;
    if (i < n) o[i] = (_Float16)w[i];
}

// ---------------------------------------------------------------------------
// Fragment loaders (ISA 7.12.2 16-bit layouts, wave32):
// lane l: row = l&15, kh = l>>4; dword d: local K = kh*8 + (d<4 ? 2d : 16+2(d-4))
// ---------------------------------------------------------------------------
__device__ inline int frag_local_k(int d, int kh) {
    return kh * 8 + ((d < 4) ? (2 * d) : (16 + 2 * (d - 4)));
}

// A fragment via implicit im2col from NHWC activations; rows in pool-group order.
template <int HW, int CINP, int KS, int PAD>
__device__ inline v16h load_A_frag(const _Float16* __restrict__ in, int b, int mtile,
                                   int chunk, int lane) {
    const int r = lane & 15;
    const int kh = lane >> 4;
    const int m = mtile * 16 + r;
    const int g = m >> 2, sub = m & 3;
    const int PW = HW / 2;
    const int Y = (g / PW) * 2 + (sub >> 1);
    const int X = (g % PW) * 2 + (sub & 1);
    Frag f;
#pragma unroll
    for (int d = 0; d < 8; ++d) {
        int k0 = chunk * 32 + frag_local_k(d, kh);
        int s = k0 / CINP;          // window position (power-of-2 CINP)
        int c = k0 % CINP;          // channel (even -> dword-aligned pair)
        unsigned v = 0u;
        if (s < KS * KS) {
            int y = Y + s / KS - PAD;
            int x = X + s % KS - PAD;
            if (y >= 0 && y < HW && x >= 0 && x < HW) {
                v = *(const unsigned*)(in + ((((size_t)b * HW + y) * HW + x) * CINP + c));
            }
        }
        f.u[d] = v;
    }
    return f.h;
}

// B fragment from N-major repacked weights Wt[n][Kpad].
__device__ inline v16h load_B_frag(const _Float16* __restrict__ Wt, int Kpad, int ntile,
                                   int chunk, int lane) {
    const int n = ntile * 16 + (lane & 15);
    const int kh = lane >> 4;
    Frag f;
#pragma unroll
    for (int d = 0; d < 8; ++d) {
        f.u[d] = *(const unsigned*)(Wt + (size_t)n * Kpad + chunk * 32 + frag_local_k(d, kh));
    }
    return f.h;
}

// ---------------------------------------------------------------------------
// Conv (implicit GEMM, WMMA f16->f32) + fused PReLU + 2x2 avg-pool.
// One block per batch; 8 waves sweep the (Mtile,Ntile) grid.
// FLAT=false: out NHWC [b][py][px][COUT]; FLAT=true: out [b][n*(M/4)+g] (NCHW flatten)
// ---------------------------------------------------------------------------
template <int HW, int CINP, int COUT, int KS, int PAD, bool FLAT>
__global__ __launch_bounds__(256) void conv_pool_kernel(const _Float16* __restrict__ in,
                                                        const _Float16* __restrict__ Wt,
                                                        const float* __restrict__ bias,
                                                        const float* __restrict__ alpha_p,
                                                        _Float16* __restrict__ out) {
    const int b = blockIdx.x;
    const int tid = threadIdx.x;
    const int wave = tid >> 5;
    const int lane = tid & 31;
    const float alpha = *alpha_p;
    constexpr int M = HW * HW;
    constexpr int MT = M / 16;
    constexpr int NT = COUT / 16;
    constexpr int CH = (KS * KS * CINP + 31) / 32;
    constexpr int Kpad = CH * 32;
    constexpr int PW = HW / 2;
    constexpr int PG = M / 4;  // pooled positions

    for (int t = wave; t < MT * NT; t += 8) {
        const int mtile = t / NT;
        const int ntile = t % NT;
        const int n = ntile * 16 + (lane & 15);
        const float bv = bias[n];
        v8f acc;
#pragma unroll
        for (int i = 0; i < 8; ++i) acc[i] = bv;
        for (int chunk = 0; chunk < CH; ++chunk) {
            v16h af = load_A_frag<HW, CINP, KS, PAD>(in, b, mtile, chunk, lane);
            v16h bf = load_B_frag(Wt, Kpad, ntile, chunk, lane);
            acc = __builtin_amdgcn_wmma_f32_16x16x32_f16(false, af, false, bf,
                                                         (short)0, acc, false, false);
        }
        float p[8];
#pragma unroll
        for (int i = 0; i < 8; ++i) {
            float v = acc[i];
            p[i] = (v >= 0.f) ? v : alpha * v;
        }
        const float q0 = (p[0] + p[1] + p[2] + p[3]) * 0.25f;
        const float q1 = (p[4] + p[5] + p[6] + p[7]) * 0.25f;
        const int mbase = mtile * 16 + (lane >> 4) * 8;
        const int g0 = mbase >> 2, g1 = g0 + 1;
        if (FLAT) {
            out[(size_t)b * (COUT * PG) + n * PG + g0] = (_Float16)q0;
            out[(size_t)b * (COUT * PG) + n * PG + g1] = (_Float16)q1;
        } else {
            const int py0 = g0 / PW, px0 = g0 % PW;
            const int py1 = g1 / PW, px1 = g1 % PW;
            out[(((size_t)b * PW + py0) * PW + px0) * COUT + n] = (_Float16)q0;
            out[(((size_t)b * PW + py1) * PW + px1) * COUT + n] = (_Float16)q1;
        }
    }
}

// ---------------------------------------------------------------------------
// Fully-connected WMMA GEMM: A[M][K] f16, Wt[N][K] f16, bias f32, optional PReLU.
// grid.x = M/16 (one M-tile per block); 8 waves sweep N-tiles.
// ---------------------------------------------------------------------------
template <int N, int K, bool PRELU, bool OUTF32>
__global__ __launch_bounds__(256) void fc_kernel(const _Float16* __restrict__ A,
                                                 const _Float16* __restrict__ Wt,
                                                 const float* __restrict__ bias,
                                                 const float* __restrict__ alpha_p,
                                                 void* __restrict__ outv) {
    const int tid = threadIdx.x;
    const int wave = tid >> 5;
    const int lane = tid & 31;
    const int r = lane & 15;
    const int kh = lane >> 4;
    const int mtile = blockIdx.x;
    const int m = mtile * 16 + r;
    constexpr int NT = N / 16;
    constexpr int CH = K / 32;
    float alpha = 0.f;
    if (PRELU) alpha = *alpha_p;

    for (int ntile = wave; ntile < NT; ntile += 8) {
        const int n = ntile * 16 + r;
        const float bv = bias[n];
        v8f acc;
#pragma unroll
        for (int i = 0; i < 8; ++i) acc[i] = bv;
        for (int chunk = 0; chunk < CH; ++chunk) {
            Frag fa, fb;
#pragma unroll
            for (int d = 0; d < 8; ++d) {
                const int k0 = chunk * 32 + frag_local_k(d, kh);
                fa.u[d] = *(const unsigned*)(A + (size_t)m * K + k0);
                fb.u[d] = *(const unsigned*)(Wt + (size_t)n * K + k0);
            }
            acc = __builtin_amdgcn_wmma_f32_16x16x32_f16(false, fa.h, false, fb.h,
                                                         (short)0, acc, false, false);
        }
        const int mb = mtile * 16 + kh * 8;
#pragma unroll
        for (int i = 0; i < 8; ++i) {
            float v = acc[i];
            if (PRELU) v = (v >= 0.f) ? v : alpha * v;
            const size_t idx = (size_t)(mb + i) * N + n;
            if (OUTF32) ((float*)outv)[idx] = v;
            else        ((_Float16*)outv)[idx] = (_Float16)v;
        }
    }
}

// ---------------------------------------------------------------------------
// Launch
// ---------------------------------------------------------------------------
extern "C" void kernel_launch(void* const* d_in, const int* in_sizes, int n_in,
                              void* d_out, int out_size, void* d_ws, size_t ws_size,
                              hipStream_t stream) {
    (void)in_sizes; (void)n_in; (void)out_size; (void)ws_size;
    const float* x   = (const float*)d_in[0];
    const float* cw1 = (const float*)d_in[1];
    const float* cb1 = (const float*)d_in[2];
    const float* a1  = (const float*)d_in[3];
    const float* cw2 = (const float*)d_in[4];
    const float* cb2 = (const float*)d_in[5];
    const float* a2  = (const float*)d_in[6];
    const float* cw3 = (const float*)d_in[7];
    const float* cb3 = (const float*)d_in[8];
    const float* a3  = (const float*)d_in[9];
    const float* fw1 = (const float*)d_in[10];
    const float* fb1 = (const float*)d_in[11];
    const float* a4  = (const float*)d_in[12];
    const float* fw2 = (const float*)d_in[13];
    const float* fb2 = (const float*)d_in[14];

    const int B = 4096;
    char* ws = (char*)d_ws;
    size_t off = 0;
    auto take = [&](size_t bytes) -> void* {
        void* p = ws + off;
        off += (bytes + 255) & ~(size_t)255;
        return p;
    };
    _Float16* d0  = (_Float16*)take((size_t)B * 32 * 32 * 4 * 2);   // hist NHWC C=4
    _Float16* d1c = (_Float16*)take((size_t)B * 16 * 16 * 32 * 2);  // after conv1+pool
    _Float16* d2c = (_Float16*)take((size_t)B * 8 * 8 * 64 * 2);    // after conv2+pool
    _Float16* d3c = (_Float16*)take((size_t)B * 2048 * 2);          // after conv3+pool (flat)
    _Float16* d4c = (_Float16*)take((size_t)B * 256 * 2);           // after FC1
    _Float16* w1  = (_Float16*)take((size_t)32 * 64 * 2);
    _Float16* w2  = (_Float16*)take((size_t)64 * 288 * 2);
    _Float16* w3  = (_Float16*)take((size_t)128 * 1600 * 2);
    _Float16* wf1 = (_Float16*)take((size_t)256 * 2048 * 2);
    _Float16* wf2 = (_Float16*)take((size_t)128 * 256 * 2);

    hist_kernel<<<B, 256, 0, stream>>>(x, d0);
    repack_conv<32, 3, 4, 3, 64><<<(32 * 64 + 255) / 256, 256, 0, stream>>>(cw1, w1);
    repack_conv<64, 32, 32, 3, 288><<<(64 * 288 + 255) / 256, 256, 0, stream>>>(cw2, w2);
    repack_conv<128, 64, 64, 5, 1600><<<(128 * 1600 + 255) / 256, 256, 0, stream>>>(cw3, w3);
    cast_f32_f16<<<(256 * 2048 + 255) / 256, 256, 0, stream>>>(fw1, wf1, 256 * 2048);
    cast_f32_f16<<<(128 * 256 + 255) / 256, 256, 0, stream>>>(fw2, wf2, 128 * 256);

    conv_pool_kernel<32, 4, 32, 3, 1, false><<<B, 256, 0, stream>>>(d0, w1, cb1, a1, d1c);
    conv_pool_kernel<16, 32, 64, 3, 1, false><<<B, 256, 0, stream>>>(d1c, w2, cb2, a2, d2c);
    conv_pool_kernel<8, 64, 128, 5, 2, true><<<B, 256, 0, stream>>>(d2c, w3, cb3, a3, d3c);

    fc_kernel<256, 2048, true, false><<<B / 16, 256, 0, stream>>>(d3c, wf1, fb1, a4, (void*)d4c);
    fc_kernel<128, 256, false, true><<<B / 16, 256, 0, stream>>>(d4c, wf2, fb2, nullptr, d_out);
}
